// MultiHeadAttention_52553219833907
// MI455X (gfx1250) — matrix operand order repaired and checked by reading, not compiled, not run on hardware
//
#include <hip/hip_runtime.h>
#include <hip/hip_bf16.h>
#include <stdint.h>

#define NHEADS 8
#define DMODEL 512
#define DHEAD  64
#define BATCH  8
#define SEQ    2048

typedef __attribute__((ext_vector_type(16))) __bf16 v16bf;
typedef __attribute__((ext_vector_type(8)))  float  v8f;
typedef __attribute__((ext_vector_type(4)))  unsigned int v4u;
typedef __attribute__((ext_vector_type(8)))  int    v8i;
typedef __attribute__((ext_vector_type(4)))  int    v4i;

// ---------------- CDNA5 async copy: global -> LDS (ASYNCcnt tracked) --------
__device__ __forceinline__ void async_copy_b128(uint32_t lds_byte_off, const void* gptr) {
  asm volatile("global_load_async_to_lds_b128 %0, %1, off"
               :: "v"(lds_byte_off), "v"(gptr)
               : "memory");
}
__device__ __forceinline__ void wait_async() {
  asm volatile("s_wait_asynccnt 0x0" ::: "memory");
}

// ---------------- CDNA5 Tensor Data Mover: 2-D tile global -> LDS -----------
// Builds a D# descriptor (groups 0/1; groups 2/3 zero for 2-D) and issues
// TENSOR_LOAD_TO_LDS. Rows of tile_d0 elements (bf16) are padded in LDS with
// 4 DWORDs after every 32 DWORDs (128B data -> 144B pitch = 72 elements).
#if __has_builtin(__builtin_amdgcn_tensor_load_to_lds)
#define HAVE_TDM 1
__device__ __forceinline__ void tdm_load_2d_bf16(uint32_t lds_off, const void* gaddr,
                                                 uint32_t tensor_d0, uint32_t tensor_d1,
                                                 uint32_t tile_d0, uint32_t tile_d1,
                                                 uint32_t stride0) {
  const uint64_t ga = (uint64_t)(uintptr_t)gaddr;
  v4u g0;
  g0[0] = 1u;                                                   // count=1, user descriptor
  g0[1] = lds_off;                                              // LDS byte address
  g0[2] = (uint32_t)ga;                                         // global_addr[31:0]
  g0[3] = (uint32_t)((ga >> 32) & 0x01FFFFFFu) | 0x80000000u;   // global_addr[56:32] | type=2
  v8i g1;
  g1[0] = (int)((1u << 16)      // data_size = 2 bytes
              | (1u << 20)      // pad_enable
              | (4u << 22)      // pad_interval: after 32 DWORDs
              | (3u << 25));    // pad_amount: 4 DWORDs
  g1[1] = (int)(tensor_d0 << 16);                               // tensor_dim0[15:0]
  g1[2] = (int)((tensor_d0 >> 16) | (tensor_d1 << 16));         // dim0 hi | dim1 lo
  g1[3] = (int)((tensor_d1 >> 16) | (tile_d0 << 16));           // dim1 hi | tile_dim0
  g1[4] = (int)(tile_d1 & 0xFFFFu);                             // tile_dim1 (tile_dim2=0)
  g1[5] = (int)stride0;                                         // tensor_dim0_stride lo
  g1[6] = 0;
  g1[7] = 0;                                                    // dim1_stride unused (2-D)
  const v4i z4 = {0, 0, 0, 0};
#if defined(__clang_major__) && (__clang_major__ >= 23)
  const v8i z8 = {0, 0, 0, 0, 0, 0, 0, 0};
  __builtin_amdgcn_tensor_load_to_lds(g0, g1, z4, z4, z8, 0);
#else
  __builtin_amdgcn_tensor_load_to_lds(g0, g1, z4, z4, 0);
#endif
}
__device__ __forceinline__ void wait_tensor() {
#if __has_builtin(__builtin_amdgcn_s_wait_tensorcnt)
  __builtin_amdgcn_s_wait_tensorcnt(0);
#else
  asm volatile("s_wait_tensorcnt 0x0" ::: "memory");
#endif
}
#endif

// ---------------- WMMA fragment helpers (bf16 16x16x32, f32 accum) ----------
// A (16x32): lanes 0-15 hold row M=lane, K = 0..15;
//            lanes 16-31 hold row M=lane-16, K = 16..31.
template <typename T>
__device__ __forceinline__ v16bf load_a_frag(const T* base, int rstride, int lane) {
  const int r  = lane & 15;
  const int kb = (lane < 16) ? 0 : 16;
  const T* p = base + (size_t)r * rstride + kb;
  v16bf a;
#pragma unroll
  for (int i = 0; i < 8; ++i) {
    a[i]     = (__bf16)p[i];
    a[i + 8] = (__bf16)p[i + 8];
  }
  return a;
}

// B (32x16): lanes 0-15 hold col N=lane, K=0..15; lanes 16-31 col N=lane-16, K=16..31.
// Source stored [col][k] (k contiguous per column).
template <typename T>
__device__ __forceinline__ v16bf load_b_kcontig(const T* src, int rstride, int lane) {
  const int c  = lane & 15;
  const int kr = (lane < 16) ? 0 : 16;
  const T* p = src + (size_t)c * rstride + kr;
  v16bf b;
#pragma unroll
  for (int i = 0; i < 16; ++i) b[i] = (__bf16)p[i];
  return b;
}

// Source stored [k][col] (k strided by pitch).
template <typename T>
__device__ __forceinline__ v16bf load_b_kstride(const T* src, int pitch, int lane) {
  const int c  = lane & 15;
  const int kr = (lane < 16) ? 0 : 16;
  const T* p = src + (size_t)kr * pitch + c;
  v16bf b;
#pragma unroll
  for (int i = 0; i < 16; ++i) b[i] = (__bf16)p[(size_t)i * pitch];
  return b;
}

__device__ __forceinline__ v8f wmma_bf16(v16bf a, v16bf b, v8f c) {
  return __builtin_amdgcn_wmma_f32_16x16x32_bf16(false, a, false, b, (short)0, c, false, false);
}

__device__ __forceinline__ v8f zero8() {
  v8f z;
#pragma unroll
  for (int i = 0; i < 8; ++i) z[i] = 0.0f;
  return z;
}

// ---------------- Projection GEMM: Y = W * X + b -----------------------------
// MODE 0: RoPE, store bf16 [b][h][n][d]   (Q and K)
// MODE 1: no RoPE, store bf16 [b][h][d][n] (V)
// MODE 2: no RoPE, store fp32 [b][o][n]    (output projection; X is bf16)
// Block: 64 output rows x 256 cols, one batch. 256 threads = 8 waves.
template <int MODE, typename XT>
__global__ __launch_bounds__(256) void proj_kernel(
    const XT* __restrict__ X, const float* __restrict__ W, const float* __restrict__ bias,
    const float* __restrict__ cosb, const float* __restrict__ sinb,
    __bf16* __restrict__ out_bf, float* __restrict__ out_f)
{
  constexpr int NT = 256;
  constexpr int PITCH_E = (sizeof(XT) == 4) ? 260 : 264;   // elems per LDS row (16B aligned, bank-spread)
  constexpr int PITCH_B = PITCH_E * (int)sizeof(XT);
  __shared__ XT xlds[32 * PITCH_E];

  const int bid = blockIdx.x;
  const int nb = bid & 7;
  const int mb = (bid >> 3) & 7;
  const int b  = bid >> 6;
  const int o0 = mb * 64, n0 = nb * NT;

  const int tid = threadIdx.x, wave = tid >> 5, lane = tid & 31;
  const int mt = wave & 3;              // m-tile (16 rows) for this wave
  const int ng = (wave >> 2) * 8;       // first n-tile index (0 or 8)
  const int half = lane >> 4, c = lane & 15;

  v8f acc[8];
#pragma unroll
  for (int j = 0; j < 8; ++j) acc[j] = zero8();

  const int cpr   = (NT * (int)sizeof(XT)) / 16;  // 16B chunks per row
  const int total = 32 * cpr;

  for (int k0 = 0; k0 < DMODEL; k0 += 32) {
    __syncthreads();
    for (int ch = tid; ch < total; ch += 256) {
      const int row = ch / cpr, coff = (ch % cpr) * 16;
      const char* g = (const char*)(X + ((size_t)b * DMODEL + k0 + row) * SEQ + n0) + coff;
      const uint32_t l = (uint32_t)(uintptr_t)((const char*)xlds + row * PITCH_B + coff);
      async_copy_b128(l, g);
    }
    wait_async();
    __syncthreads();

    const v16bf aw = load_a_frag(W + (size_t)(o0 + 16 * mt) * DMODEL + k0, DMODEL, lane);
#pragma unroll
    for (int j = 0; j < 8; ++j) {
      const v16bf bx = load_b_kstride(xlds + (ng + j) * 16, PITCH_E, lane);
      acc[j] = wmma_bf16(aw, bx, acc[j]);
    }
  }

  // epilogue: bias (+ RoPE) + layout-transforming store
  float bv[8];
#pragma unroll
  for (int r = 0; r < 8; ++r) bv[r] = bias[o0 + 16 * mt + r + 8 * half];
  const int dloc = (o0 + 16 * mt) / 8 + half;     // d index for this lane (same for all 8 VGPRs)

#pragma unroll
  for (int j = 0; j < 8; ++j) {
    const int n = n0 + (ng + j) * 16 + c;
    if (MODE == 0) {
      const float cv = cosb[(size_t)dloc * SEQ + n];
      const float sv = sinb[(size_t)dloc * SEQ + n];
#pragma unroll
      for (int r = 0; r < 8; ++r) {
        const float x = acc[j][r] + bv[r];
        const float other = __shfl_xor(x, 16, 32);     // RoPE pair is lane^16, same VGPR
        const float rot = half ? other : -other;
        const float y = x * cv + rot * sv;
        out_bf[(((size_t)b * NHEADS + r) * SEQ + n) * DHEAD + dloc] = (__bf16)y;  // h == r
      }
    } else if (MODE == 1) {
#pragma unroll
      for (int r = 0; r < 8; ++r) {
        const float x = acc[j][r] + bv[r];
        out_bf[(((size_t)b * NHEADS + r) * DHEAD + dloc) * SEQ + n] = (__bf16)x;
      }
    } else {
#pragma unroll
      for (int r = 0; r < 8; ++r) {
        const int o = o0 + 16 * mt + r + 8 * half;
        out_f[((size_t)b * DMODEL + o) * SEQ + n] = acc[j][r] + bv[r];
      }
    }
  }
}

// ---------------- Flash attention ------------------------------------------
// Block = one (b,h) + 128 queries; 8 waves * 16 query rows each.
// KV streamed in 64-key chunks into LDS via the Tensor Data Mover (one wave
// issues two 2-D tile descriptors with LDS row padding; TENSORcnt + barrier).
__global__ __launch_bounds__(256) void attn_kernel(
    const __bf16* __restrict__ Q, const __bf16* __restrict__ K, const __bf16* __restrict__ V,
    __bf16* __restrict__ ctx)
{
  constexpr int KB = 64;    // keys per chunk
  constexpr int KP = 72;    // K LDS pitch (elems): 144B rows (TDM pad: 4DW per 32DW)
  constexpr int VP = 72;    // V LDS pitch
  constexpr int PP = 72;    // P LDS pitch
  __shared__ __bf16 klds[KB * KP];          // [m][d]
  __shared__ __bf16 vlds[DHEAD * VP];       // [d][m]
  __shared__ __bf16 plds[8 * 16 * PP];      // per-wave P tiles [n][m]

  const int bid = blockIdx.x;               // B*H*(SEQ/128)
  const int qb = bid & 15;
  const int h  = (bid >> 4) & 7;
  const int b  = bid >> 7;
  const int q0 = qb * 128;

  const int tid = threadIdx.x, wave = tid >> 5, lane = tid & 31;
  const int half = lane >> 4, c = lane & 15;

  const __bf16* Qh = Q + ((size_t)b * NHEADS + h) * SEQ * DHEAD;   // [n][d]
  const __bf16* Kh = K + ((size_t)b * NHEADS + h) * SEQ * DHEAD;   // [m][d]
  const __bf16* Vh = V + ((size_t)b * NHEADS + h) * DHEAD * SEQ;   // [d][m]

  const int nq = q0 + wave * 16;
  const v16bf aq0 = load_a_frag(Qh + (size_t)nq * DHEAD,      DHEAD, lane);  // d 0..31
  const v16bf aq1 = load_a_frag(Qh + (size_t)nq * DHEAD + 32, DHEAD, lane);  // d 32..63

  float mrow[8], lrow[8];
  v8f oacc[4];
#pragma unroll
  for (int r = 0; r < 8; ++r) { mrow[r] = -1e30f; lrow[r] = 0.0f; }
#pragma unroll
  for (int dt = 0; dt < 4; ++dt) oacc[dt] = zero8();

  const float SCL = 0.125f * 1.44269504f;   // 1/sqrt(dh) * log2(e)
  __bf16* pw = plds + (size_t)wave * 16 * PP;

  for (int m0 = 0; m0 < SEQ; m0 += KB) {
    __syncthreads();
#if defined(HAVE_TDM)
    if (wave == 0) {
      // K tile: tensor [SEQ][64], tile 64 rows(m) x 64 elems(d), stride 64
      tdm_load_2d_bf16((uint32_t)(uintptr_t)klds, Kh + (size_t)m0 * DHEAD,
                       DHEAD, SEQ, DHEAD, KB, DHEAD);
      // V tile: tensor [64][SEQ], tile 64 rows(d) x 64 elems(m), stride SEQ
      tdm_load_2d_bf16((uint32_t)(uintptr_t)vlds, Vh + m0,
                       SEQ, DHEAD, KB, DHEAD, SEQ);
      wait_tensor();
    }
#else
    for (int ch = tid; ch < 512; ch += 256) {
      const int row = ch >> 3, off = (ch & 7) * 16;
      async_copy_b128((uint32_t)(uintptr_t)((const char*)klds + row * (KP * 2) + off),
                      (const char*)(Kh + (size_t)(m0 + row) * DHEAD) + off);
    }
    for (int ch = tid; ch < 512; ch += 256) {
      const int row = ch >> 3, off = (ch & 7) * 16;
      async_copy_b128((uint32_t)(uintptr_t)((const char*)vlds + row * (VP * 2) + off),
                      (const char*)(Vh + (size_t)row * SEQ + m0) + off);
    }
    wait_async();
#endif
    __syncthreads();

    // S = Q * K^T  (4 tiles of 16x16, K-dim = dh = 64 in two WMMA steps)
    v8f st[4];
#pragma unroll
    for (int t = 0; t < 4; ++t) {
      v8f z = zero8();
      const v16bf b0 = load_b_kcontig(klds + (t * 16) * KP,      KP, lane);  // d 0..31
      const v16bf b1 = load_b_kcontig(klds + (t * 16) * KP + 32, KP, lane);  // d 32..63
      z = wmma_bf16(aq0, b0, z);
      z = wmma_bf16(aq1, b1, z);
      st[t] = z;
    }

    // online softmax (row = r + 8*half, columns across 16 lanes of the half)
    float fs[8];
#pragma unroll
    for (int r = 0; r < 8; ++r) {
      float mx = st[0][r];
#pragma unroll
      for (int t = 1; t < 4; ++t) mx = fmaxf(mx, st[t][r]);
      mx = fmaxf(mx, __shfl_xor(mx, 1, 32));
      mx = fmaxf(mx, __shfl_xor(mx, 2, 32));
      mx = fmaxf(mx, __shfl_xor(mx, 4, 32));
      mx = fmaxf(mx, __shfl_xor(mx, 8, 32));
      const float mnew = fmaxf(mrow[r], mx);
      fs[r] = exp2f((mrow[r] - mnew) * SCL);
      mrow[r] = mnew;
      lrow[r] *= fs[r];
    }
#pragma unroll
    for (int dt = 0; dt < 4; ++dt)
#pragma unroll
      for (int r = 0; r < 8; ++r) oacc[dt][r] *= fs[r];

    float ps[8];
#pragma unroll
    for (int r = 0; r < 8; ++r) ps[r] = 0.0f;
#pragma unroll
    for (int t = 0; t < 4; ++t) {
#pragma unroll
      for (int r = 0; r < 8; ++r) {
        const float p = exp2f((st[t][r] - mrow[r]) * SCL);
        ps[r] += p;
        pw[(r + 8 * half) * PP + t * 16 + c] = (__bf16)p;   // P tile [n][m] in LDS
      }
    }
#pragma unroll
    for (int r = 0; r < 8; ++r) {
      float s = ps[r];
      s += __shfl_xor(s, 1, 32);
      s += __shfl_xor(s, 2, 32);
      s += __shfl_xor(s, 4, 32);
      s += __shfl_xor(s, 8, 32);
      lrow[r] += s;
    }

    // O += P * V   (A = P 16x64 from this wave's LDS; B = V^T from shared LDS)
#pragma unroll
    for (int s = 0; s < 2; ++s) {
      const v16bf pa = load_a_frag(pw + s * 32, PP, lane);
#pragma unroll
      for (int dt = 0; dt < 4; ++dt) {
        const v16bf bvf = load_b_kcontig(vlds + (dt * 16) * VP + s * 32, VP, lane);
        oacc[dt] = wmma_bf16(pa, bvf, oacc[dt]);
      }
    }
  }

  // finalize: O/l, store ctx bf16 as [b][i = d*H + h][n] for the output GEMM
  float inv[8];
#pragma unroll
  for (int r = 0; r < 8; ++r) inv[r] = (lrow[r] > 0.0f) ? 1.0f / lrow[r] : 0.0f;
#pragma unroll
  for (int dt = 0; dt < 4; ++dt) {
    const int d = dt * 16 + c;
#pragma unroll
    for (int r = 0; r < 8; ++r) {
      const int n = nq + r + 8 * half;
      const int i = d * NHEADS + h;
      ctx[((size_t)b * DMODEL + i) * SEQ + n] = (__bf16)(oacc[dt][r] * inv[r]);
    }
  }
}

// ---------------- launcher ---------------------------------------------------
extern "C" void kernel_launch(void* const* d_in, const int* in_sizes, int n_in,
                              void* d_out, int out_size, void* d_ws, size_t ws_size,
                              hipStream_t stream) {
  const float* query = (const float*)d_in[0];
  const float* key   = (const float*)d_in[1];
  const float* value = (const float*)d_in[2];
  const float* cosb  = (const float*)d_in[3];
  const float* sinb  = (const float*)d_in[4];
  const float* wq = (const float*)d_in[5];  const float* bq = (const float*)d_in[6];
  const float* wk = (const float*)d_in[7];  const float* bk = (const float*)d_in[8];
  const float* wv = (const float*)d_in[9];  const float* bv = (const float*)d_in[10];
  const float* wm = (const float*)d_in[11]; const float* bm = (const float*)d_in[12];
  float* out = (float*)d_out;

  const size_t hsz = (size_t)BATCH * NHEADS * SEQ * DHEAD;  // elements per q/k/v buffer
  __bf16* qb = reinterpret_cast<__bf16*>(d_ws);
  __bf16* kb = qb + hsz;
  __bf16* vb = kb + hsz;
  __bf16* cb = vb + hsz;   // ctx, bf16 [b][D][n]; total ws use = 4*hsz*2 = 64 MiB

  const dim3 blk(256);
  const dim3 gP(BATCH * (DMODEL / 64) * (SEQ / 256));       // 512 blocks
  const dim3 gA(BATCH * NHEADS * (SEQ / 128));              // 1024 blocks

  proj_kernel<0, float><<<gP, blk, 0, stream>>>(query, wq, bq, cosb, sinb, qb, nullptr);
  proj_kernel<0, float><<<gP, blk, 0, stream>>>(key,   wk, bk, cosb, sinb, kb, nullptr);
  proj_kernel<1, float><<<gP, blk, 0, stream>>>(value, wv, bv, nullptr, nullptr, vb, nullptr);
  attn_kernel<<<gA, blk, 0, stream>>>(qb, kb, vb, cb);
  proj_kernel<2, __bf16><<<gP, blk, 0, stream>>>(cb, wm, bm, nullptr, nullptr, nullptr, out);
}
